// GCNLayer_62886911148372
// MI455X (gfx1250) — matrix operand order repaired and checked by reading, not compiled, not run on hardware
//
#include <hip/hip_runtime.h>
#include <hip/hip_bf16.h>
#include <stdint.h>

// ---- types -----------------------------------------------------------------
typedef __bf16 bf16_t;
typedef __attribute__((ext_vector_type(16))) __bf16 v16bf;
typedef __attribute__((ext_vector_type(2)))  __bf16 bf16x2;
typedef __attribute__((ext_vector_type(8)))  float  v8f;
typedef __attribute__((ext_vector_type(4)))  unsigned u32x4;
typedef __attribute__((ext_vector_type(2)))  unsigned u32x2;
typedef __attribute__((ext_vector_type(4)))  float  f32x4;
typedef __attribute__((ext_vector_type(2)))  float  f32x2;

#define B_DIM 16
#define C_DIM 64
#define S_DIM 2048
#define F_DIM 64
#define TS     8      // s-values per block == waves per block
#define PITCH  72     // LDS row pitch (halves) for X / A / W  (16B aligned rows)
#define SPITCH 64     // LDS row pitch (halves) for support

// fp32 pair -> packed bf16x2 (RNE). __builtin_convertvector lowers to
// fptrunc <2 x float> -> <2 x bfloat>, selectable as one v_cvt_pk_bf16_f32.
__device__ __forceinline__ unsigned pack_bf16(float lo, float hi) {
  f32x2 v;
  v.x = lo;
  v.y = hi;
  bf16x2 p = __builtin_convertvector(v, bf16x2);
  return __builtin_bit_cast(unsigned, p);
}

__device__ __forceinline__ bf16_t to_bf16(float f) { return (bf16_t)f; }

union FragU { v16bf v; u32x4 q[2]; };

// A-operand (16x32 bf16): lane L -> row r=L%16; group g=L/16 selects K halves
// {g*8+0..7, g*8+16..23} (ISA 16-bit A layout). Row-major storage.
__device__ __forceinline__ v16bf load_a_frag(const bf16_t* base, int pitch,
                                             int r, int g, int koff) {
  const bf16_t* row = base + r * pitch + koff + g * 8;
  FragU f;
  f.q[0] = *(const u32x4*)(row);
  f.q[1] = *(const u32x4*)(row + 16);
  return f.v;
}

// B-operand (32x16 bf16): lane L -> column n=L%16; group g selects K halves
// {g*8+0..7, g*8+16..23}. Storage is transposed ([n][K]) so reads are contiguous.
__device__ __forceinline__ v16bf load_b_frag(const bf16_t* base, int pitch,
                                             int n, int g, int koff) {
  const bf16_t* row = base + n * pitch + koff + g * 8;
  FragU f;
  f.q[0] = *(const u32x4*)(row);
  f.q[1] = *(const u32x4*)(row + 16);
  return f.v;
}

__global__ __launch_bounds__(256)
void gcn_fused_kernel(const float* __restrict__ x,
                      const float* __restrict__ adj,
                      const float* __restrict__ weight,
                      float* __restrict__ out) {
  __shared__ __align__(16) bf16_t X_lds[TS * C_DIM * PITCH];      // [s][d][f]   72 KB
  __shared__ __align__(16) bf16_t A_lds[C_DIM * PITCH];           // [c][d]       9 KB
  __shared__ __align__(16) bf16_t Wt_lds[C_DIM * PITCH];          // [o][f]       9 KB
  __shared__ __align__(16) bf16_t Supp_lds[TS * C_DIM * SPITCH];  // per-wave [o][d] 64 KB
  __shared__ float part[256];
  __shared__ float rs[C_DIM];

  const int t    = threadIdx.x;
  const int lane = t & 31;
  const int w    = t >> 5;        // wave id == local s index
  const int r    = lane & 15;
  const int g    = lane >> 4;

  const int blk = blockIdx.x;
  const int b   = blk >> 8;              // 256 s-chunks per batch
  const int s0  = (blk & 255) * TS;

  // ---- degree rsqrt: 4 threads per row, 16 elements each (adj is L2-resident)
  {
    const int row = t >> 2, q = t & 3;
    float sum = 0.f;
    #pragma unroll
    for (int d = 0; d < 16; ++d) sum += adj[row * C_DIM + q * 16 + d];
    part[t] = sum;
  }
  __syncthreads();
  if (t < C_DIM)
    rs[t] = rsqrtf(part[4 * t] + part[4 * t + 1] + part[4 * t + 2] + part[4 * t + 3]);
  __syncthreads();

  // ---- normalized adjacency + transposed weight -> LDS (bf16)
  for (int idx = t; idx < C_DIM * C_DIM; idx += 256) {
    int row = idx >> 6, col = idx & 63;
    A_lds[row * PITCH + col]  = to_bf16(adj[idx] * rs[row] * rs[col]);
    // weight[f][o] -> Wt[o][f]
    Wt_lds[col * PITCH + row] = to_bf16(weight[idx]);
  }

  // ---- stream X chunk: x[b, d, s0..s0+7, :]; per d the 512 floats are contiguous.
  // Each iteration covers two d-rows; each thread does one b128 load + b64 DS store.
  const float* xb = x + ((size_t)b * C_DIM) * (S_DIM * F_DIM) + (size_t)s0 * F_DIM;
  {
    const int dd  = t >> 7;            // which of the two d-rows this thread serves
    const int idx = (t & 127) * 4;     // element within the 512-float row-group
    const int sl  = idx >> 6;          // local s
    const int f   = idx & 63;
    const unsigned dsoff = (unsigned)((sl * C_DIM) * PITCH + f);
    for (int d0 = 0; d0 < C_DIM; d0 += 2) {
      const int d = d0 + dd;
      if (d + 8 < C_DIM)
        __builtin_prefetch(xb + (size_t)(d + 8) * (S_DIM * F_DIM) + idx);
      const f32x4 v = *(const f32x4*)(xb + (size_t)d * (S_DIM * F_DIM) + idx);
      u32x2 pk;
      pk.x = pack_bf16(v.x, v.y);
      pk.y = pack_bf16(v.z, v.w);
      *(u32x2*)(X_lds + dsoff + d * PITCH) = pk;
    }
  }
  __syncthreads();

  const bf16_t* Xw = X_lds    + (w * C_DIM) * PITCH;   // this wave's s-slice
  bf16_t*       Sw = Supp_lds + w * C_DIM * SPITCH;    // this wave's support slab

  // ---- GEMM1: supp[d][o] = sum_f X[d][f] * W[f][o]; store transposed [o][d]
  for (int dt = 0; dt < 4; ++dt) {
    const v16bf a0 = load_a_frag(Xw + dt * 16 * PITCH, PITCH, r, g, 0);
    const v16bf a1 = load_a_frag(Xw + dt * 16 * PITCH, PITCH, r, g, 32);
    #pragma unroll
    for (int ot = 0; ot < 4; ++ot) {
      v8f acc = {};
      v16bf b0 = load_b_frag(Wt_lds + ot * 16 * PITCH, PITCH, r, g, 0);
      acc = __builtin_amdgcn_wmma_f32_16x16x32_bf16(false, a0, false, b0,
                                                    (short)0, acc, false, false);
      v16bf b1 = load_b_frag(Wt_lds + ot * 16 * PITCH, PITCH, r, g, 32);
      acc = __builtin_amdgcn_wmma_f32_16x16x32_bf16(false, a1, false, b1,
                                                    (short)0, acc, false, false);
      u32x4 p;
      p.x = pack_bf16(acc[0], acc[1]);
      p.y = pack_bf16(acc[2], acc[3]);
      p.z = pack_bf16(acc[4], acc[5]);
      p.w = pack_bf16(acc[6], acc[7]);
      // D-frag: (d = dt*16 + g*8 + i, o = ot*16 + r) -> Supp[o][d], 8 halves = b128
      *(u32x4*)(Sw + (ot * 16 + r) * SPITCH + dt * 16 + g * 8) = p;
    }
  }

  // ---- GEMM2: out[c][o] = relu( sum_d A_norm[c][d] * supp[d][o] )
  float* ob = out + (((size_t)b * C_DIM) * S_DIM + (size_t)(s0 + w)) * F_DIM;
  for (int ct = 0; ct < 4; ++ct) {
    const v16bf a0 = load_a_frag(A_lds + ct * 16 * PITCH, PITCH, r, g, 0);
    const v16bf a1 = load_a_frag(A_lds + ct * 16 * PITCH, PITCH, r, g, 32);
    #pragma unroll
    for (int ot = 0; ot < 4; ++ot) {
      v8f acc = {};
      v16bf b0 = load_b_frag(Sw + ot * 16 * SPITCH, SPITCH, r, g, 0);
      acc = __builtin_amdgcn_wmma_f32_16x16x32_bf16(false, a0, false, b0,
                                                    (short)0, acc, false, false);
      v16bf b1 = load_b_frag(Sw + ot * 16 * SPITCH, SPITCH, r, g, 32);
      acc = __builtin_amdgcn_wmma_f32_16x16x32_bf16(false, a1, false, b1,
                                                    (short)0, acc, false, false);
      #pragma unroll
      for (int i = 0; i < 8; ++i) {
        float v = acc[i] > 0.f ? acc[i] : 0.f;
        int c = ct * 16 + g * 8 + i;
        int o = ot * 16 + r;
        ob[(size_t)c * (S_DIM * F_DIM) + o] = v;
      }
    }
  }
}

extern "C" void kernel_launch(void* const* d_in, const int* in_sizes, int n_in,
                              void* d_out, int out_size, void* d_ws, size_t ws_size,
                              hipStream_t stream) {
  (void)in_sizes; (void)n_in; (void)out_size; (void)d_ws; (void)ws_size;
  const float* x      = (const float*)d_in[0];
  const float* adj    = (const float*)d_in[1];
  const float* weight = (const float*)d_in[2];
  float* out          = (float*)d_out;

  dim3 grid(B_DIM * (S_DIM / TS));   // 16 * 256 = 4096 blocks
  dim3 block(256);                   // 8 wave32 waves
  gcn_fused_kernel<<<grid, block, 0, stream>>>(x, adj, weight, out);
}
